// HierarchicalAttention_46815143526595
// MI455X (gfx1250) — matrix operand order repaired and checked by reading, not compiled, not run on hardware
//
#include <hip/hip_runtime.h>
#include <hip/hip_bf16.h>

typedef __bf16 bf16;
typedef __attribute__((ext_vector_type(16))) __bf16 v16bf;
typedef __attribute__((ext_vector_type(8)))  __bf16 v8bf;
typedef __attribute__((ext_vector_type(8)))  float  v8f;
typedef __attribute__((ext_vector_type(4)))  float  v4f;

#define WMMA_BF16(a, b, c) \
  __builtin_amdgcn_wmma_f32_16x16x32_bf16(false, (a), false, (b), (short)0, (c), false, false)

union V16 { v16bf v; v8bf h[2]; };

constexpr int BB  = 4;      // batch
constexpr int NN  = 8192;   // sequence
constexpr int DD  = 1024;   // model dim == H*DH
constexpr int HH  = 16;     // heads
constexpr int DHD = 64;     // head dim
constexpr int CCB = 32;     // block size C
constexpr int NT  = BB * NN;  // 32768 tokens

// ---------------------------------------------------------------------------
// Weight convert + transpose: W[k][n] fp32 -> Wt[n][k] bf16  (K=N=1024)
// ---------------------------------------------------------------------------
__global__ __launch_bounds__(256) void wconv_kernel(const float* __restrict__ W,
                                                    bf16* __restrict__ Wt) {
  int idx = blockIdx.x * 256 + threadIdx.x;   // 1M elements
  int k = idx >> 10, n = idx & 1023;
  Wt[(long)n * DD + k] = (bf16)W[idx];
}

// ---------------------------------------------------------------------------
// Tiled bf16 transpose: dst[c][t] = src[t][c]  (c dim = DD, t dim = ntok)
// grid: (ntok/32, 32), 256 threads
// ---------------------------------------------------------------------------
__global__ __launch_bounds__(256) void vtrans_kernel(const bf16* __restrict__ src,
                                                     bf16* __restrict__ dst, int ntok) {
  __shared__ bf16 tile[32][33];
  int t0 = blockIdx.x * 32, c0 = blockIdx.y * 32;
  int tx = threadIdx.x & 31, ty = threadIdx.x >> 5;   // ty in [0,8)
#pragma unroll
  for (int j = 0; j < 32; j += 8)
    tile[ty + j][tx] = src[(long)(t0 + ty + j) * DD + c0 + tx];
  __syncthreads();
#pragma unroll
  for (int j = 0; j < 32; j += 8)
    dst[(long)(c0 + ty + j) * ntok + t0 + tx] = tile[tx][ty + j];
}

// ---------------------------------------------------------------------------
// QKV projection: Y = (X @ W) * scale + bias  (per-token mask on V)
// one wave -> 32x64 output tile; K-loop 1024/32
// grid: (2048, 3) x 256 threads
// ---------------------------------------------------------------------------
__global__ __launch_bounds__(256) void proj_in_kernel(
    const float* __restrict__ Xq, const float* __restrict__ Xkv,
    const bf16* __restrict__ WqT, const bf16* __restrict__ WkT, const bf16* __restrict__ WvT,
    const float* __restrict__ bq, const float* __restrict__ bk, const float* __restrict__ bv,
    const int* __restrict__ kmask,
    bf16* __restrict__ q0, bf16* __restrict__ k0, bf16* __restrict__ v0) {
  const int mat = blockIdx.y;
  const float* X    = (mat == 0) ? Xq : Xkv;
  const bf16*  Wt   = (mat == 0) ? WqT : (mat == 1 ? WkT : WvT);
  const float* bias = (mat == 0) ? bq : (mat == 1 ? bk : bv);
  bf16*        Y    = (mat == 0) ? q0 : (mat == 1 ? k0 : v0);
  const float scale = (mat == 0) ? 0.125f : 1.0f;   // 1/sqrt(64) folded into q

  int wid  = blockIdx.x * 8 + (threadIdx.x >> 5);   // 16384 wave-tiles
  int lane = threadIdx.x & 31;
  int tm = wid >> 4, tn = wid & 15;                 // tm: 32-row tile, tn: 64-col tile
  int m = lane & 15, kg = lane >> 4;

  v8f acc[2][4] = {};
  for (int kk = 0; kk < DD; kk += 32) {
    V16 A[2];
#pragma unroll
    for (int mt = 0; mt < 2; mt++) {
      const float* ap = X + (long)(tm * 32 + mt * 16 + m) * DD + kk;
      __builtin_prefetch(ap + 32, 0, 1);
      v4f f0 = *(const v4f*)(ap + kg * 8);
      v4f f1 = *(const v4f*)(ap + kg * 8 + 4);
      v4f f2 = *(const v4f*)(ap + 16 + kg * 8);
      v4f f3 = *(const v4f*)(ap + 16 + kg * 8 + 4);
#pragma unroll
      for (int i = 0; i < 4; i++) {
        A[mt].v[i]      = (bf16)f0[i];
        A[mt].v[4 + i]  = (bf16)f1[i];
        A[mt].v[8 + i]  = (bf16)f2[i];
        A[mt].v[12 + i] = (bf16)f3[i];
      }
    }
#pragma unroll
    for (int s = 0; s < 4; s++) {
      int col = tn * 64 + s * 16 + m;
      V16 Bv;
      Bv.h[0] = *(const v8bf*)(Wt + (long)col * DD + kk + kg * 8);
      Bv.h[1] = *(const v8bf*)(Wt + (long)col * DD + kk + 16 + kg * 8);
      acc[0][s] = WMMA_BF16(A[0].v, Bv.v, acc[0][s]);
      acc[1][s] = WMMA_BF16(A[1].v, Bv.v, acc[1][s]);
    }
  }
#pragma unroll
  for (int mt = 0; mt < 2; mt++) {
#pragma unroll
    for (int s = 0; s < 4; s++) {
      int col = tn * 64 + s * 16 + m;
      float bval = bias[col];
#pragma unroll
      for (int r = 0; r < 8; r++) {
        int row = tm * 32 + mt * 16 + r + 8 * kg;
        float val = acc[mt][s][r] * scale + bval;
        if (mat == 2) val *= (float)kmask[row];   // v = v * key_mask
        Y[(long)row * DD + col] = (bf16)val;
      }
    }
  }
}

// ---------------------------------------------------------------------------
// Output projection: out = attn(bf16) @ Wo + bo   (fp32 out), 32x64 per wave
// ---------------------------------------------------------------------------
__global__ __launch_bounds__(256) void proj_out_kernel(
    const bf16* __restrict__ A, const bf16* __restrict__ WoT,
    const float* __restrict__ bo, float* __restrict__ out) {
  int wid  = blockIdx.x * 8 + (threadIdx.x >> 5);
  int lane = threadIdx.x & 31;
  int tm = wid >> 4, tn = wid & 15;
  int m = lane & 15, kg = lane >> 4;

  v8f acc[2][4] = {};
  for (int kk = 0; kk < DD; kk += 32) {
    V16 Av[2];
#pragma unroll
    for (int mt = 0; mt < 2; mt++) {
      const bf16* ap = A + (long)(tm * 32 + mt * 16 + m) * DD + kk;
      __builtin_prefetch(ap + 32, 0, 1);
      Av[mt].h[0] = *(const v8bf*)(ap + kg * 8);
      Av[mt].h[1] = *(const v8bf*)(ap + 16 + kg * 8);
    }
#pragma unroll
    for (int s = 0; s < 4; s++) {
      int col = tn * 64 + s * 16 + m;
      V16 Bv;
      Bv.h[0] = *(const v8bf*)(WoT + (long)col * DD + kk + kg * 8);
      Bv.h[1] = *(const v8bf*)(WoT + (long)col * DD + kk + 16 + kg * 8);
      acc[0][s] = WMMA_BF16(Av[0].v, Bv.v, acc[0][s]);
      acc[1][s] = WMMA_BF16(Av[1].v, Bv.v, acc[1][s]);
    }
  }
#pragma unroll
  for (int mt = 0; mt < 2; mt++) {
#pragma unroll
    for (int s = 0; s < 4; s++) {
      int col = tn * 64 + s * 16 + m;
      float bval = bo[col];
#pragma unroll
      for (int r = 0; r < 8; r++) {
        int row = tm * 32 + mt * 16 + r + 8 * kg;
        out[(long)row * DD + col] = acc[mt][s][r] + bval;
      }
    }
  }
}

// ---------------------------------------------------------------------------
// Coarsening: level-l row i = mean of 2^l contiguous level-0 rows
// ---------------------------------------------------------------------------
__global__ __launch_bounds__(256) void pool_kernel(
    const bf16* __restrict__ q0, const bf16* __restrict__ k0, const bf16* __restrict__ v0,
    bf16* __restrict__ qd, bf16* __restrict__ kd, bf16* __restrict__ vd,
    int lvl, int nl) {
  int row = blockIdx.x;               // b*nl + i
  int b = row / nl, i = row % nl;
  int span = 1 << lvl;
  long base = ((long)b * NN + (long)i * span) * DD;
  float inv = 1.0f / (float)span;
  for (int c = threadIdx.x; c < DD; c += 256) {
    float sq = 0.f, sk = 0.f, sv = 0.f;
    for (int j = 0; j < span; j++) {
      long o = base + (long)j * DD + c;
      sq += (float)q0[o]; sk += (float)k0[o]; sv += (float)v0[o];
    }
    long d = (long)row * DD + c;
    qd[d] = (bf16)(sq * inv); kd[d] = (bf16)(sk * inv); vd[d] = (bf16)(sv * inv);
  }
}

__global__ void mask_pool_kernel(const int* __restrict__ m0, float* __restrict__ md,
                                 int lvl, int nl) {
  int idx = blockIdx.x * blockDim.x + threadIdx.x;   // b*nl + i
  if (idx >= BB * nl) return;
  int b = idx / nl, i = idx % nl;
  int span = 1 << lvl;
  float s = 0.f;
  for (int j = 0; j < span; j++) s += (float)m0[b * NN + i * span + j];
  md[idx] = s / (float)span;
}

// ---------------------------------------------------------------------------
// Per-level block attention. One wave handles one head of one (b, blk).
// vT is channel-major V: vT[c][b*nl + t]  -> contiguous WMMA B-frag loads.
// grid: (BB*nb, 2) x 256 threads (8 waves = 8 heads per block)
// ---------------------------------------------------------------------------
__global__ __launch_bounds__(256) void attn_level_kernel(
    const bf16* __restrict__ qL, const bf16* __restrict__ kL, const bf16* __restrict__ vT,
    const float* __restrict__ mL, const int* __restrict__ m0, const float* __restrict__ rpb,
    float* __restrict__ numO, float* __restrict__ denO, int lvl, int nl, int nb) {
  __shared__ bf16  lds_w[8][32 * 104];   // per-wave w [32 q][96 p], stride 104
  __shared__ float lds_rpb[8][128];

  int wave = threadIdx.x >> 5;
  int lane = threadIdx.x & 31;
  int b   = blockIdx.x / nb;
  int blk = blockIdx.x % nb;
  int h   = blockIdx.y * 8 + wave;
  int m = lane & 15, kg = lane >> 4;
  long qbase = (long)b * nl + blk * CCB;
  long ntok = (long)BB * nl;

  // preload this head's rpb row (127 entries)
  {
    long base = ((long)lvl * HH + h) * 127;
    for (int i = lane; i < 127; i += 32) lds_rpb[wave][i] = rpb[base + i];
  }

  // ---- logits = q @ k^T : 32x96, K=64 ----
  v8f Cf[2][6] = {};
#pragma unroll
  for (int kt = 0; kt < 2; kt++) {
    V16 Af[2];
#pragma unroll
    for (int mt = 0; mt < 2; mt++) {
      const bf16* ap = qL + (qbase + mt * 16 + m) * DD + h * DHD + kt * 32;
      Af[mt].h[0] = *(const v8bf*)(ap + kg * 8);
      Af[mt].h[1] = *(const v8bf*)(ap + 16 + kg * 8);
    }
#pragma unroll
    for (int nt = 0; nt < 6; nt++) {
      int wb = blk - 1 + (nt >> 1);          // window block (uniform)
      if (wb < 0 || wb >= nb) continue;      // zero-padded edge: skip (C stays 0)
      int pl = (nt & 1) * 16 + m;            // key offset inside block
      const bf16* bp = kL + ((long)b * nl + wb * CCB + pl) * DD + h * DHD + kt * 32 + kg * 8;
      V16 Bv;
      Bv.h[0] = *(const v8bf*)(bp);
      Bv.h[1] = *(const v8bf*)(bp + 16);
      Cf[0][nt] = WMMA_BF16(Af[0].v, Bv.v, Cf[0][nt]);
      Cf[1][nt] = WMMA_BF16(Af[1].v, Bv.v, Cf[1][nt]);
    }
  }

  // ---- epilogue: w = exp(logit + rpb) * valid * key_mask -> LDS (bf16) ----
#pragma unroll
  for (int mt = 0; mt < 2; mt++) {
#pragma unroll
    for (int nt = 0; nt < 6; nt++) {
      int p    = nt * 16 + m;
      int pblk = nt >> 1;
      int pl   = (nt & 1) * 16 + m;
      int wb   = blk - 1 + pblk;
      float mw = 0.0f;
      if (wb >= 0 && wb < nb) {
        long kr = (long)b * nl + wb * CCB + pl;
        mw = (lvl == 0) ? (float)m0[kr] : mL[kr];
      }
#pragma unroll
      for (int r = 0; r < 8; r++) {
        int qrow = mt * 16 + r + 8 * kg;
        float lg = Cf[mt][nt][r] + lds_rpb[wave][p - qrow + 31];
        bool inval = (lvl > 0) && ((pblk == 1) ||
                     (pblk == 0 && qrow < 16 && pl >= 16) ||
                     (pblk == 2 && qrow >= 16 && pl < 16));
        float w = inval ? 0.0f : __expf(lg) * mw;
        lds_w[wave][qrow * 104 + p] = (bf16)w;
      }
    }
  }

  // ---- den: per-query row sums (lane == query row, 32 rows) ----
  {
    float s = 0.0f;
    const bf16* wr = &lds_w[wave][lane * 104];
    for (int p = 0; p < 96; p++) s += (float)wr[p];
    denO[((long)b * nl + blk * CCB + lane) * HH + h] = s;
  }

  // ---- num = w @ vw : 32x64, K=96; B frags contiguous from vT ----
  v8f Nf[2][4] = {};
#pragma unroll
  for (int ks = 0; ks < 3; ks++) {
    int wb = blk - 1 + ks;                   // uniform per ks
    if (wb < 0 || wb >= nb) continue;
    V16 A2[2];
#pragma unroll
    for (int mt = 0; mt < 2; mt++) {
      const bf16* ap = &lds_w[wave][(mt * 16 + m) * 104 + ks * 32];
      A2[mt].h[0] = *(const v8bf*)(ap + kg * 8);
      A2[mt].h[1] = *(const v8bf*)(ap + 16 + kg * 8);
    }
    int koff = kg * 8;
#pragma unroll
    for (int nt = 0; nt < 4; nt++) {
      int c = h * DHD + nt * 16 + m;         // channel (dh column)
      const bf16* vp = vT + (long)c * ntok + (long)b * nl + wb * CCB + koff;
      V16 Bv;
      Bv.h[0] = *(const v8bf*)(vp);
      Bv.h[1] = *(const v8bf*)(vp + 16);
      Nf[0][nt] = WMMA_BF16(A2[0].v, Bv.v, Nf[0][nt]);
      Nf[1][nt] = WMMA_BF16(A2[1].v, Bv.v, Nf[1][nt]);
    }
  }

#pragma unroll
  for (int mt = 0; mt < 2; mt++)
#pragma unroll
    for (int nt = 0; nt < 4; nt++)
#pragma unroll
      for (int r = 0; r < 8; r++) {
        int qrow = mt * 16 + r + 8 * kg;
        numO[(((long)b * nl + blk * CCB + qrow) * HH + h) * DHD + nt * 16 + m] = Nf[mt][nt][r];
      }
}

// ---------------------------------------------------------------------------
// Gather: sum levels (nearest-neighbor upsample), normalize, apply qmask
// ---------------------------------------------------------------------------
__global__ __launch_bounds__(256) void gather_kernel(
    const float* __restrict__ numL, const float* __restrict__ denL,
    const int* __restrict__ qmask, bf16* __restrict__ attn) {
  long row = blockIdx.x;                    // b*N + n
  int b = (int)(row / NN), n = (int)(row % NN);
  float qm = (float)qmask[row];
  for (int c = threadIdx.x; c < DD; c += 256) {
    int h = c >> 6;
    float s = 0.0f, d = 0.0f;
#pragma unroll
    for (int l = 0; l < 8; l++) {
      int nl = NN >> l;
      long off = (l == 0) ? 0 : (long)BB * (2 * NN - (NN >> (l - 1)));
      long tok = off + (long)b * nl + (n >> l);
      s += numL[tok * DD + c];
      d += denL[tok * HH + h];
    }
    attn[row * DD + c] = (bf16)(qm * s / (d + 1e-9f));
  }
}

// ---------------------------------------------------------------------------
extern "C" void kernel_launch(void* const* d_in, const int* in_sizes, int n_in,
                              void* d_out, int out_size, void* d_ws, size_t ws_size,
                              hipStream_t stream) {
  const float* Xq    = (const float*)d_in[0];
  const float* Xkv   = (const float*)d_in[1];
  const int*   qmask = (const int*)d_in[2];
  const int*   kmask = (const int*)d_in[3];
  const float* Wq = (const float*)d_in[4];
  const float* bq = (const float*)d_in[5];
  const float* Wk = (const float*)d_in[6];
  const float* bk = (const float*)d_in[7];
  const float* Wv = (const float*)d_in[8];
  const float* bv = (const float*)d_in[9];
  const float* Wo = (const float*)d_in[10];
  const float* bo = (const float*)d_in[11];
  const float* rpb = (const float*)d_in[12];
  float* out = (float*)d_out;

  char* wsb = (char*)d_ws;
  size_t off = 0;
  auto walloc = [&](size_t bytes) -> void* {
    void* p = wsb + off;
    off = (off + bytes + 255) & ~(size_t)255;
    return p;
  };
  bf16* q0 = (bf16*)walloc((size_t)NT * DD * 2);
  bf16* k0 = (bf16*)walloc((size_t)NT * DD * 2);
  bf16* v0 = (bf16*)walloc((size_t)NT * DD * 2);
  size_t pyrTok = (size_t)BB * (NN - (NN >> 7));        // tokens, levels 1..7
  bf16* pq = (bf16*)walloc(pyrTok * DD * 2);
  bf16* pk = (bf16*)walloc(pyrTok * DD * 2);
  bf16* pv = (bf16*)walloc(pyrTok * DD * 2);
  float* pm = (float*)walloc(pyrTok * 4);
  bf16* vT0 = (bf16*)walloc((size_t)NT * DD * 2);       // channel-major V, level 0
  bf16* vTp = (bf16*)walloc(pyrTok * DD * 2);           // channel-major V, levels 1..7
  size_t allTok = (size_t)BB * (2 * NN - (NN >> 7));    // tokens, levels 0..7
  float* numL = (float*)walloc(allTok * DD * 4);
  float* denL = (float*)walloc(allTok * HH * 4);
  bf16* attn = (bf16*)walloc((size_t)NT * DD * 2);
  bf16* WqT = (bf16*)walloc((size_t)DD * DD * 2);
  bf16* WkT = (bf16*)walloc((size_t)DD * DD * 2);
  bf16* WvT = (bf16*)walloc((size_t)DD * DD * 2);
  bf16* WoT = (bf16*)walloc((size_t)DD * DD * 2);

  // 1) weight convert/transpose to bf16 [N][K]
  wconv_kernel<<<4096, 256, 0, stream>>>(Wq, WqT);
  wconv_kernel<<<4096, 256, 0, stream>>>(Wk, WkT);
  wconv_kernel<<<4096, 256, 0, stream>>>(Wv, WvT);
  wconv_kernel<<<4096, 256, 0, stream>>>(Wo, WoT);

  // 2) q/k/v projections (WMMA bf16), 32x64 tile per wave
  proj_in_kernel<<<dim3(2048, 3), 256, 0, stream>>>(
      Xq, Xkv, WqT, WkT, WvT, bq, bk, bv, kmask, q0, k0, v0);

  // 3) pyramids (levels 1..7)
  for (int l = 1; l <= 7; l++) {
    int nl = NN >> l;
    size_t po = (size_t)BB * (NN - (NN >> (l - 1)));
    pool_kernel<<<BB * nl, 256, 0, stream>>>(
        q0, k0, v0, pq + po * DD, pk + po * DD, pv + po * DD, l, nl);
    mask_pool_kernel<<<(BB * nl + 255) / 256, 256, 0, stream>>>(kmask, pm + po, l, nl);
  }

  // 3b) channel-major V per level (for contiguous WMMA B-frag loads)
  vtrans_kernel<<<dim3(NT / 32, 32), 256, 0, stream>>>(v0, vT0, NT);
  for (int l = 1; l <= 7; l++) {
    int nl = NN >> l;
    size_t po = (size_t)BB * (NN - (NN >> (l - 1)));
    vtrans_kernel<<<dim3(BB * nl / 32, 32), 256, 0, stream>>>(
        pv + po * DD, vTp + po * DD, BB * nl);
  }

  // 4) per-level block attention (WMMA bf16)
  for (int l = 0; l < 8; l++) {
    int nl = NN >> l;
    int nb = nl / CCB;
    size_t po = (l == 0) ? 0 : (size_t)BB * (NN - (NN >> (l - 1)));
    size_t to = (l == 0) ? 0 : (size_t)BB * (2 * NN - (NN >> (l - 1)));
    const bf16* ql = (l == 0) ? q0 : pq + po * DD;
    const bf16* kl = (l == 0) ? k0 : pk + po * DD;
    const bf16* vt = (l == 0) ? vT0 : vTp + po * DD;
    const float* ml = (l == 0) ? nullptr : pm + po;
    attn_level_kernel<<<dim3(BB * nb, 2), 256, 0, stream>>>(
        ql, kl, vt, ml, kmask, rpb, numL + to * DD, denL + to * HH, l, nl, nb);
  }

  // 5) upsample-sum + normalize + qmask -> bf16 attn
  gather_kernel<<<NT, 256, 0, stream>>>(numL, denL, qmask, attn);

  // 6) output projection (WMMA bf16, fp32 out)
  proj_out_kernel<<<2048, 256, 0, stream>>>(attn, WoT, bo, out);
}
